// MPNNModel_15401752723912
// MI455X (gfx1250) — matrix operand order repaired and checked by reading, not compiled, not run on hardware
//
#include <hip/hip_runtime.h>
#include <hip/hip_bf16.h>

// MPNN: edge MLP (WMMA bf16) -> u_mul_e + scatter-add -> node MLP (WMMA bf16)
// -> per-graph softmax attention pooling -> classifier.
// Heavy GEMMs use v_wmma_f32_16x16x32_bf16 (f32 accumulate). Memory-bound by
// the edge gather/scatter (~1.2 GB @ 23.3 TB/s), so bf16 matrix math keeps
// compute off the critical path.

#define NN 100000
#define EE 1600000
#define GG 128
#define DD 64
#define CC 10

typedef __attribute__((ext_vector_type(16))) __bf16 v16bf;
typedef __attribute__((ext_vector_type(8)))  float  v8f;
typedef __attribute__((ext_vector_type(4)))  unsigned int v4u;

union Frag {
    v16bf v;
    v4u   q[2];
};

__device__ __forceinline__ v8f wmma_bf16(v16bf a, v16bf b, v8f c) {
    return __builtin_amdgcn_wmma_f32_16x16x32_bf16(false, a, false, b, (short)0, c,
                                                   false, false);
}

// ---------------------------------------------------------------- converts
__global__ __launch_bounds__(256) void cvt_x_kernel(const float* __restrict__ x,
                                                    __bf16* __restrict__ hbf) {
    int i = blockIdx.x * 256 + threadIdx.x;
    if (i < NN * DD) hbf[i] = (__bf16)x[i];
}

// Pack We1 [128,64], We2 [64,64], Wn [64,64] into WMMA B-fragment order:
// fragment f covers K = kk*32..+31, N = nt*16..+15 (f = kk*4 + nt).
// Element (lane, i): K = kk*32 + (i>=8 ? 16 : 0) + (lane>=16 ? 8 : 0) + (i&7) ; N = nt*16 + (lane&15).
__global__ __launch_bounds__(256) void pack_w_kernel(const float* __restrict__ We1,
                                                     const float* __restrict__ We2,
                                                     const float* __restrict__ Wn,
                                                     __bf16* __restrict__ we1f,
                                                     __bf16* __restrict__ we2f,
                                                     __bf16* __restrict__ wnf) {
    int t = blockIdx.x * 256 + threadIdx.x;  // 0 .. 16383
    const float* srcW;
    __bf16* dstW;
    int idx;
    if (t < 8192) { srcW = We1; dstW = we1f; idx = t; }
    else if (t < 12288) { srcW = We2; dstW = we2f; idx = t - 8192; }
    else if (t < 16384) { srcW = Wn;  dstW = wnf;  idx = t - 12288; }
    else return;
    int f = idx >> 9;          // fragment id
    int r = idx & 511;
    int lane = r >> 4;
    int i = r & 15;
    int kk = f >> 2;
    int nt = f & 3;
    int K = kk * 32 + ((i >> 3) << 4) + ((lane >> 4) << 3) + (i & 7);
    int Nc = nt * 16 + (lane & 15);
    dstW[idx] = (__bf16)srcW[K * 64 + Nc];
}

// ---------------------------------------------------------------- edge stage
// 256 threads = 8 waves; block handles 128 edges (16 per wave).
__global__ __launch_bounds__(256) void edge_kernel(const __bf16* __restrict__ hbf,
                                                   const int* __restrict__ src,
                                                   const int* __restrict__ dst,
                                                   const __bf16* __restrict__ we1f,
                                                   const __bf16* __restrict__ we2f,
                                                   const float* __restrict__ be1,
                                                   const float* __restrict__ be2,
                                                   float* __restrict__ h_neigh) {
    __shared__ __align__(16) __bf16 ein[128][136];   // concat(h_src, h_dst), +pad
    __shared__ __align__(16) __bf16 hid[8][16][72];  // per-wave hidden tile
    __shared__ int   dsts[128];
    __shared__ float b1s[64], b2s[64];

    const int tid = threadIdx.x;
    const int lane = tid & 31;
    const int wv = tid >> 5;
    const int lane15 = lane & 15;
    const int laneHi = lane >> 4;  // 0 or 1
    const int e0 = blockIdx.x * 128;

    // Stage 1: gather. thread t -> edge (t&127), src half (t<128) / dst half.
    {
        int el = tid & 127;
        int isDst = tid >> 7;
        int node = (isDst ? dst : src)[e0 + el];
        if (isDst) dsts[el] = node;
        const v4u* rowp = (const v4u*)(hbf + (size_t)node * DD);
        v4u* outp = (v4u*)&ein[el][isDst * 64];
#pragma unroll
        for (int k = 0; k < 8; ++k) outp[k] = rowp[k];
        if (tid < 64) { b1s[tid] = be1[tid]; b2s[tid] = be2[tid]; }
    }
    __syncthreads();

    const int m0 = wv * 16;
    const int r = m0 + lane15;

    v8f acc[4];
    v8f zero8 = {0.f, 0.f, 0.f, 0.f, 0.f, 0.f, 0.f, 0.f};
#pragma unroll
    for (int nt = 0; nt < 4; ++nt) acc[nt] = zero8;

    // GEMM1: [16,128] x [128,64]
#pragma unroll
    for (int kk = 0; kk < 4; ++kk) {
        const int kb = kk * 32;
        Frag fa;
        fa.q[0] = *(const v4u*)&ein[r][kb + laneHi * 8];
        fa.q[1] = *(const v4u*)&ein[r][kb + 16 + laneHi * 8];
#pragma unroll
        for (int nt = 0; nt < 4; ++nt) {
            Frag fb;
            const __bf16* bp = we1f + ((size_t)((kk * 4 + nt) * 32 + lane) * 16);
            fb.q[0] = *(const v4u*)(bp);
            fb.q[1] = *(const v4u*)(bp + 8);
            acc[nt] = wmma_bf16(fa.v, fb.v, acc[nt]);
        }
    }

    // bias + ReLU -> bf16 hidden tile (per-wave LDS region)
#pragma unroll
    for (int nt = 0; nt < 4; ++nt) {
        float bb = b1s[nt * 16 + lane15];
#pragma unroll
        for (int j = 0; j < 8; ++j) {
            float v = acc[nt][j] + bb;
            v = v > 0.f ? v : 0.f;
            hid[wv][j + laneHi * 8][nt * 16 + lane15] = (__bf16)v;
        }
        acc[nt] = zero8;
    }
    asm volatile("s_wait_dscnt 0x0" ::: "memory");  // wave-local LDS RAW fence

    // GEMM2: [16,64] x [64,64]
#pragma unroll
    for (int ks = 0; ks < 2; ++ks) {
        const int kb = ks * 32;
        Frag fa;
        fa.q[0] = *(const v4u*)&hid[wv][lane15][kb + laneHi * 8];
        fa.q[1] = *(const v4u*)&hid[wv][lane15][kb + 16 + laneHi * 8];
#pragma unroll
        for (int nt = 0; nt < 4; ++nt) {
            Frag fb;
            const __bf16* bp = we2f + ((size_t)((ks * 4 + nt) * 32 + lane) * 16);
            fb.q[0] = *(const v4u*)(bp);
            fb.q[1] = *(const v4u*)(bp + 8);
            acc[nt] = wmma_bf16(fa.v, fb.v, acc[nt]);
        }
    }

    // msg = h_src * (m + be2); scatter-add onto destination nodes
#pragma unroll
    for (int nt = 0; nt < 4; ++nt) {
        int n = nt * 16 + lane15;
        float bb = b2s[n];
#pragma unroll
        for (int j = 0; j < 8; ++j) {
            int m = j + laneHi * 8;
            float mv = acc[nt][j] + bb;
            float hs = (float)ein[m0 + m][n];  // first 64 cols == h_src
            unsafeAtomicAdd(&h_neigh[(size_t)dsts[m0 + m] * DD + n], hs * mv);
        }
    }
}

// ---------------------------------------------------------------- node stage
__global__ __launch_bounds__(256) void node_kernel(const float* __restrict__ h_neigh,
                                                   const __bf16* __restrict__ wnf,
                                                   const float* __restrict__ bn,
                                                   float* __restrict__ h2) {
    __shared__ __align__(16) __bf16 tile[8][16][72];

    const int tid = threadIdx.x;
    const int lane = tid & 31;
    const int wv = tid >> 5;
    const int lane15 = lane & 15;
    const int laneHi = lane >> 4;
    const int nb = blockIdx.x * 128 + wv * 16;  // wave-uniform

    if (nb + 16 > NN) return;  // uniform tail guard (N % 16 == 0)

    // load 16 rows of h_neigh, convert to bf16
    {
        int rr = lane15;
        int cbase = laneHi * 32;
        const float* rowp = h_neigh + (size_t)(nb + rr) * DD + cbase;
#pragma unroll
        for (int c = 0; c < 32; ++c) tile[wv][rr][cbase + c] = (__bf16)rowp[c];
    }
    asm volatile("s_wait_dscnt 0x0" ::: "memory");

    v8f acc[4];
    v8f zero8 = {0.f, 0.f, 0.f, 0.f, 0.f, 0.f, 0.f, 0.f};
#pragma unroll
    for (int nt = 0; nt < 4; ++nt) acc[nt] = zero8;

#pragma unroll
    for (int ks = 0; ks < 2; ++ks) {
        const int kb = ks * 32;
        Frag fa;
        fa.q[0] = *(const v4u*)&tile[wv][lane15][kb + laneHi * 8];
        fa.q[1] = *(const v4u*)&tile[wv][lane15][kb + 16 + laneHi * 8];
#pragma unroll
        for (int nt = 0; nt < 4; ++nt) {
            Frag fb;
            const __bf16* bp = wnf + ((size_t)((ks * 4 + nt) * 32 + lane) * 16);
            fb.q[0] = *(const v4u*)(bp);
            fb.q[1] = *(const v4u*)(bp + 8);
            acc[nt] = wmma_bf16(fa.v, fb.v, acc[nt]);
        }
    }

#pragma unroll
    for (int nt = 0; nt < 4; ++nt) {
        int n = nt * 16 + lane15;
        float bb = bn[n];
#pragma unroll
        for (int j = 0; j < 8; ++j) {
            int m = j + laneHi * 8;
            float v = acc[nt][j] + bb;
            h2[(size_t)(nb + m) * DD + n] = v > 0.f ? v : 0.f;
        }
    }
}

// ---------------------------------------------------------------- pooling
__device__ __forceinline__ int lower_bound_dev(const int* a, int n, int key) {
    int lo = 0, hi = n;
    while (lo < hi) {
        int mid = (lo + hi) >> 1;
        if (a[mid] < key) lo = mid + 1; else hi = mid;
    }
    return lo;
}

// One block per graph. graph_ids is sorted -> contiguous node range.
__global__ __launch_bounds__(256) void pool_kernel(const float* __restrict__ h2,
                                                   const float* __restrict__ Wg,
                                                   const float* __restrict__ bg,
                                                   const float* __restrict__ Wfc,
                                                   const float* __restrict__ bfc,
                                                   const int* __restrict__ gid,
                                                   float* __restrict__ gateA,
                                                   float* __restrict__ out) {
    const int g = blockIdx.x;
    const int tid = threadIdx.x;
    __shared__ float s_wg[64];
    __shared__ float s_red[256];
    __shared__ float s_pool[4][64];
    __shared__ float s_pooled[64];
    __shared__ int s_lo, s_hi;
    __shared__ float s_gmax, s_den;

    if (tid < 64) s_wg[tid] = Wg[tid];
    if (tid == 0) {
        s_lo = lower_bound_dev(gid, NN, g);
        s_hi = lower_bound_dev(gid, NN, g + 1);
    }
    __syncthreads();
    const int lo = s_lo, hi = s_hi;
    const float bgv = bg[0];

    // pass 1: gates + max
    float lmax = -3.4e38f;
    for (int n = lo + tid; n < hi; n += 256) {
        const float* hr = h2 + (size_t)n * DD;
        float s = bgv;
#pragma unroll
        for (int d = 0; d < DD; ++d) s += hr[d] * s_wg[d];
        gateA[n] = s;
        lmax = fmaxf(lmax, s);
    }
    s_red[tid] = lmax;
    __syncthreads();
    for (int st = 128; st > 0; st >>= 1) {
        if (tid < st) s_red[tid] = fmaxf(s_red[tid], s_red[tid + st]);
        __syncthreads();
    }
    if (tid == 0) s_gmax = s_red[0];
    __syncthreads();
    const float gm = s_gmax;

    // pass 2: exp + sum
    float lsum = 0.f;
    for (int n = lo + tid; n < hi; n += 256) {
        float a = __expf(gateA[n] - gm);
        gateA[n] = a;
        lsum += a;
    }
    s_red[tid] = lsum;
    __syncthreads();
    for (int st = 128; st > 0; st >>= 1) {
        if (tid < st) s_red[tid] += s_red[tid + st];
        __syncthreads();
    }
    if (tid == 0) s_den = s_red[0];
    __threadfence_block();
    __syncthreads();
    const float inv = s_den > 0.f ? 1.f / s_den : 0.f;  // empty graph -> 0

    // pass 3: pooled[d] = inv * sum_n a[n] * h2[n][d]
    const int sub = tid >> 6;
    const int d = tid & 63;
    float accd = 0.f;
    for (int n = lo + sub; n < hi; n += 4) accd += gateA[n] * h2[(size_t)n * DD + d];
    s_pool[sub][d] = accd;
    __syncthreads();
    if (tid < 64)
        s_pooled[tid] = (s_pool[0][tid] + s_pool[1][tid] + s_pool[2][tid] + s_pool[3][tid]) * inv;
    __syncthreads();

    // classifier
    if (tid < CC) {
        float s = bfc[tid];
#pragma unroll
        for (int dd = 0; dd < DD; ++dd) s += s_pooled[dd] * Wfc[dd * CC + tid];
        out[g * CC + tid] = s;
    }
}

// ---------------------------------------------------------------- launch
extern "C" void kernel_launch(void* const* d_in, const int* in_sizes, int n_in,
                              void* d_out, int out_size, void* d_ws, size_t ws_size,
                              hipStream_t stream) {
    const float* x   = (const float*)d_in[0];
    const float* We1 = (const float*)d_in[1];
    const float* be1 = (const float*)d_in[2];
    const float* We2 = (const float*)d_in[3];
    const float* be2 = (const float*)d_in[4];
    const float* Wn  = (const float*)d_in[5];
    const float* bn  = (const float*)d_in[6];
    const float* Wg  = (const float*)d_in[7];
    const float* bg  = (const float*)d_in[8];
    const float* Wfc = (const float*)d_in[9];
    const float* bfc = (const float*)d_in[10];
    const int* src = (const int*)d_in[11];
    const int* dst = (const int*)d_in[12];
    const int* gid = (const int*)d_in[13];
    float* out = (float*)d_out;

    // workspace layout (256B aligned)
    char* ws = (char*)d_ws;
    size_t off = 0;
    auto take = [&](size_t bytes) {
        char* p = ws + off;
        off = (off + bytes + 255) & ~(size_t)255;
        return p;
    };
    __bf16* hbf     = (__bf16*)take((size_t)NN * DD * 2);  // 12.8 MB
    float*  h_neigh = (float*)take((size_t)NN * DD * 4);   // 25.6 MB
    float*  h2      = (float*)take((size_t)NN * DD * 4);   // 25.6 MB
    float*  gateA   = (float*)take((size_t)NN * 4);        // 0.4 MB
    __bf16* we1f    = (__bf16*)take(16 * 512 * 2);
    __bf16* we2f    = (__bf16*)take(8 * 512 * 2);
    __bf16* wnf     = (__bf16*)take(8 * 512 * 2);
    (void)ws_size; (void)in_sizes; (void)n_in; (void)out_size;

    hipMemsetAsync(h_neigh, 0, (size_t)NN * DD * 4, stream);
    cvt_x_kernel<<<(NN * DD + 255) / 256, 256, 0, stream>>>(x, hbf);
    pack_w_kernel<<<64, 256, 0, stream>>>(We1, We2, Wn, we1f, we2f, wnf);
    edge_kernel<<<EE / 128, 256, 0, stream>>>(hbf, src, dst, we1f, we2f, be1, be2, h_neigh);
    node_kernel<<<(NN + 127) / 128, 256, 0, stream>>>(h_neigh, wnf, bn, h2);
    pool_kernel<<<GG, 256, 0, stream>>>(h2, Wg, bg, Wfc, bfc, gid, gateA, out);
}
